// GSRR_73306501808769
// MI455X (gfx1250) — matrix operand order repaired and checked by reading, not compiled, not run on hardware
//
#include <hip/hip_runtime.h>
#include <math.h>

typedef __attribute__((ext_vector_type(16))) _Float16 v16h;
typedef __attribute__((ext_vector_type(8)))  _Float16 v8h;
typedef __attribute__((ext_vector_type(8)))  float    v8f;

#define C_DIM 768
#define M_PTS 32
#define HW_SZ 676
#define B_SZ  64

// ---------------------------------------------------------------------------
// Kernel 1: pack W (f32 [768 x 1536]) -> f16, same row-major layout.
// ---------------------------------------------------------------------------
__global__ void pack_w_kernel(const float* __restrict__ W1,
                              const float* __restrict__ W2,
                              _Float16* __restrict__ Wp1,
                              _Float16* __restrict__ Wp2, int n) {
    int i = blockIdx.x * blockDim.x + threadIdx.x;
    if (i < n) {
        Wp1[i] = (_Float16)W1[i];
        Wp2[i] = (_Float16)W2[i];
    }
}

// ---------------------------------------------------------------------------
// Kernel 2: attention-weighted adaptive max pool.
// fm[b,c,m] = max_hw features[b,c,hw] * attn[b,m,hw]
// One block per batch; attn[b] (86.5 KB) cached in LDS; each thread owns a
// channel c and keeps 32 running maxima in registers (features read once).
// ---------------------------------------------------------------------------
__global__ __launch_bounds__(256) void pool_kernel(
    const float* __restrict__ features,   // [B, C, HW]
    const float* __restrict__ attn,       // [B, M, HW]
    float* __restrict__ fm)               // [B, C, M]
{
    extern __shared__ float a_s[];        // [M_PTS * HW_SZ]
    const int tid = threadIdx.x;
    const int b = blockIdx.x;

    for (int i = tid; i < M_PTS * HW_SZ; i += 256)
        a_s[i] = attn[(size_t)b * M_PTS * HW_SZ + i];
    __syncthreads();

    for (int c = tid; c < C_DIM; c += 256) {
        const float* frow = features + ((size_t)b * C_DIM + c) * HW_SZ;
        float acc[M_PTS];
#pragma unroll
        for (int m = 0; m < M_PTS; ++m) acc[m] = -INFINITY;
        for (int hw = 0; hw < HW_SZ; ++hw) {
            float fv = frow[hw];
#pragma unroll
            for (int m = 0; m < M_PTS; ++m)
                acc[m] = fmaxf(acc[m], fv * a_s[m * HW_SZ + hw]);
        }
        float* orow = fm + ((size_t)b * C_DIM + c) * M_PTS;
#pragma unroll
        for (int m = 0; m < M_PTS; ++m) orow[m] = acc[m];
    }
}

// ---------------------------------------------------------------------------
// Kernel 3: EdgeConv. One block (8 wave32) per batch.
//   1) fT = f16 transpose of f [32][768] in LDS
//   2) Gram matrix -> knn top-3 (lower index wins ties, like lax.top_k)
//   3) P = Wa*f, Q = Wb*f via v_wmma_f32_16x16x32_f16 -> f32 LDS
//   4) y[o,n,j] = P[o,idx] - P[o,n] + Q[o,n]; BN; LeakyReLU(0.2); max_j
// LDS: 48K fT + 96K P + 96K Q + 4K G + idx + BN tables = 256384 B (<320K)
// ---------------------------------------------------------------------------
#define LDS_FT     0
#define LDS_P      (49152)
#define LDS_Q      (49152 + 98304)
#define LDS_G      (49152 + 2 * 98304)
#define LDS_IDX    (LDS_G + 4096)
#define LDS_BNSC   (LDS_IDX + 384)
#define LDS_TOTAL  (LDS_BNSC + 2 * 3072)

__global__ __launch_bounds__(256) void edgeconv_kernel(
    const float* __restrict__ fin,        // [B, C, M]
    const _Float16* __restrict__ Wp,      // [C, 2C] f16 row-major
    const float* __restrict__ gamma, const float* __restrict__ beta,
    const float* __restrict__ mean,  const float* __restrict__ var,
    float* __restrict__ fout)             // [B, C, M]
{
    extern __shared__ char smem[];
    _Float16* fT   = (_Float16*)(smem + LDS_FT);   // [32][768]
    float*    P    = (float*)(smem + LDS_P);       // [768][32]
    float*    Q    = (float*)(smem + LDS_Q);       // [768][32]
    float*    G    = (float*)(smem + LDS_G);       // [32][32]
    int*      nidx = (int*)(smem + LDS_IDX);       // [32][3]
    float*    bnsc = (float*)(smem + LDS_BNSC);    // [768]
    float*    bnsh = bnsc + C_DIM;                 // [768]

    const int tid = threadIdx.x;
    const int b = blockIdx.x;
    const float* f = fin + (size_t)b * C_DIM * M_PTS;

    // BN folded scale/shift
    for (int o = tid; o < C_DIM; o += 256) {
        float sc = gamma[o] * rsqrtf(var[o] + 1e-5f);
        bnsc[o] = sc;
        bnsh[o] = beta[o] - mean[o] * sc;
    }
    // f -> f16 transpose in LDS (coalesced global reads)
    for (int i = tid; i < C_DIM * M_PTS; i += 256) {
        int c = i >> 5, n = i & 31;
        fT[n * C_DIM + c] = (_Float16)f[i];
    }
    __syncthreads();

    // Gram matrix G[n][m] = <f_n, f_m>
    for (int p = tid; p < M_PTS * M_PTS; p += 256) {
        int n = p >> 5, m = p & 31;
        const _Float16* fn = fT + n * C_DIM;
        const _Float16* fmv = fT + m * C_DIM;
        float s = 0.f;
        for (int c = 0; c < C_DIM; ++c) s += (float)fn[c] * (float)fmv[c];
        G[p] = s;
    }
    __syncthreads();

    // top-3 of pd[n][m] = -|f_n|^2 + 2 G - |f_m|^2 (stable: strict >)
    if (tid < M_PTS) {
        int n = tid;
        float xn = G[n * 32 + n];
        float v0 = -INFINITY, v1 = -INFINITY, v2 = -INFINITY;
        int i0 = 0, i1 = 0, i2 = 0;
        for (int m = 0; m < M_PTS; ++m) {
            float pd = 2.f * G[n * 32 + m] - xn - G[m * 32 + m];
            if (pd > v0)      { v2 = v1; i2 = i1; v1 = v0; i1 = i0; v0 = pd; i0 = m; }
            else if (pd > v1) { v2 = v1; i2 = i1; v1 = pd; i1 = m; }
            else if (pd > v2) { v2 = pd; i2 = m; }
        }
        nidx[n * 3 + 0] = i0; nidx[n * 3 + 1] = i1; nidx[n * 3 + 2] = i2;
    }
    __syncthreads();

    // ---- WMMA GEMM: P = Wa * f , Q = Wb * f ------------------------------
    // A (16x32, MxK): lane holds row o=lane%16; element e -> K = (e/8)*16
    //                 + (lane/16)*8 + e%8  => two contiguous 8-half runs.
    // B (32x16, KxN): lane holds col n=lane%16; element e -> K = (e/8)*16
    //                 + (lane/16)*8 + e%8  => two contiguous 8-half runs of fT[n][*].
    const int wave = tid >> 5;
    const int lane = tid & 31;
    const int laneLo = lane & 15;
    const int laneHi = lane >> 4;
    union V16 { v16h v; v8h h[2]; };

    for (int mt = wave; mt < C_DIM / 16; mt += 8) {
        const int orow = mt * 16 + laneLo;
        for (int half = 0; half < 2; ++half) {
            v8f acc0 = {};
            v8f acc1 = {};
            const _Float16* wrow = Wp + (size_t)orow * (2 * C_DIM) + half * C_DIM;
            for (int kc = 0; kc < C_DIM; kc += 32) {
                V16 a, b0, b1;
                const v8h* pa = (const v8h*)(wrow + kc + laneHi * 8);
                a.h[0] = pa[0];               // K = kc + laneHi*8 + [0,8)
                a.h[1] = pa[2];               // K = kc + 16 + laneHi*8 + [0,8)
                const v8h* pb0 = (const v8h*)(fT + (size_t)laneLo * C_DIM + kc + laneHi * 8);
                b0.h[0] = pb0[0]; b0.h[1] = pb0[2];
                const v8h* pb1 = (const v8h*)(fT + (size_t)(16 + laneLo) * C_DIM + kc + laneHi * 8);
                b1.h[0] = pb1[0]; b1.h[1] = pb1[2];
                acc0 = __builtin_amdgcn_wmma_f32_16x16x32_f16(
                    false, a.v, false, b0.v, (short)0, acc0, false, false);
                acc1 = __builtin_amdgcn_wmma_f32_16x16x32_f16(
                    false, a.v, false, b1.v, (short)0, acc1, false, false);
            }
            float* dst = half ? Q : P;
#pragma unroll
            for (int v = 0; v < 8; ++v) {     // D: row = v + 8*(lane/16), col = lane%16
                int row = mt * 16 + v + 8 * laneHi;
                dst[row * 32 + laneLo]      = acc0[v];
                dst[row * 32 + 16 + laneLo] = acc1[v];
            }
        }
    }
    __syncthreads();

    // ---- epilogue: gather, BN, LeakyReLU, max over k ---------------------
    float* out = fout + (size_t)b * C_DIM * M_PTS;
    for (int i = tid; i < C_DIM * M_PTS; i += 256) {
        int o = i >> 5, n = i & 31;
        float pself = P[o * 32 + n];
        float q = Q[o * 32 + n];
        float sc = bnsc[o], sh = bnsh[o];
        float best = -INFINITY;
#pragma unroll
        for (int j = 0; j < 3; ++j) {
            float y = P[o * 32 + nidx[n * 3 + j]] - pself + q;
            y = y * sc + sh;
            y = (y >= 0.f) ? y : 0.2f * y;
            best = fmaxf(best, y);
        }
        out[i] = best;
    }
}

// ---------------------------------------------------------------------------
// Kernel 4: fuse = [f1|f2] flat; signed-sqrt; L2 normalize per batch row.
// ---------------------------------------------------------------------------
__global__ __launch_bounds__(256) void finalize_kernel(
    const float* __restrict__ f1, const float* __restrict__ f2,
    float* __restrict__ out)
{
    __shared__ float red[256];
    const int tid = threadIdx.x;
    const int b = blockIdx.x;
    const int CM = C_DIM * M_PTS;          // 24576
    const float* r1 = f1 + (size_t)b * CM;
    const float* r2 = f2 + (size_t)b * CM;

    float ss = 0.f;
    for (int i = tid; i < 2 * CM; i += 256) {
        float x = (i < CM) ? r1[i] : r2[i - CM];
        float sg = (x > 0.f) ? 1.f : ((x < 0.f) ? -1.f : 0.f);
        float s = sg * sqrtf(fabsf(x) + 1e-12f);
        ss += s * s;
    }
    red[tid] = ss;
    __syncthreads();
    for (int st = 128; st > 0; st >>= 1) {
        if (tid < st) red[tid] += red[tid + st];
        __syncthreads();
    }
    float inv = 1.f / fmaxf(sqrtf(red[0]), 1e-12f);
    for (int i = tid; i < 2 * CM; i += 256) {
        float x = (i < CM) ? r1[i] : r2[i - CM];
        float sg = (x > 0.f) ? 1.f : ((x < 0.f) ? -1.f : 0.f);
        float s = sg * sqrtf(fabsf(x) + 1e-12f);
        out[(size_t)b * 2 * CM + i] = s * inv;
    }
}

// ---------------------------------------------------------------------------
extern "C" void kernel_launch(void* const* d_in, const int* in_sizes, int n_in,
                              void* d_out, int out_size, void* d_ws, size_t ws_size,
                              hipStream_t stream) {
    (void)in_sizes; (void)n_in; (void)out_size; (void)ws_size;
    const float* features = (const float*)d_in[0];
    const float* attn     = (const float*)d_in[1];
    const float* W1       = (const float*)d_in[2];
    const float* g1       = (const float*)d_in[3];
    const float* be1      = (const float*)d_in[4];
    const float* m1       = (const float*)d_in[5];
    const float* v1       = (const float*)d_in[6];
    const float* W2       = (const float*)d_in[7];
    const float* g2       = (const float*)d_in[8];
    const float* be2      = (const float*)d_in[9];
    const float* m2       = (const float*)d_in[10];
    const float* v2       = (const float*)d_in[11];

    const size_t wpk_bytes = (size_t)C_DIM * 2 * C_DIM * sizeof(_Float16); // 2,359,296
    const size_t fm_elems  = (size_t)B_SZ * C_DIM * M_PTS;                 // 1,572,864

    char* ws = (char*)d_ws;
    _Float16* Wp1 = (_Float16*)ws;
    _Float16* Wp2 = (_Float16*)(ws + wpk_bytes);
    float* fm = (float*)(ws + 2 * wpk_bytes);
    float* f1 = fm + fm_elems;
    float* f2 = f1 + fm_elems;

    const int nw = C_DIM * 2 * C_DIM;
    pack_w_kernel<<<(nw + 255) / 256, 256, 0, stream>>>(W1, W2, Wp1, Wp2, nw);

    pool_kernel<<<B_SZ, 256, M_PTS * HW_SZ * sizeof(float), stream>>>(
        features, attn, fm);

    edgeconv_kernel<<<B_SZ, 256, LDS_TOTAL, stream>>>(
        fm, Wp1, g1, be1, m1, v1, f1);
    edgeconv_kernel<<<B_SZ, 256, LDS_TOTAL, stream>>>(
        f1, Wp2, g2, be2, m2, v2, f2);

    finalize_kernel<<<B_SZ, 256, 0, stream>>>(f1, f2, (float*)d_out);
}
